// PerformerBlock_13975823582173
// MI455X (gfx1250) — matrix operand order repaired and checked by reading, not compiled, not run on hardware
//
#include <hip/hip_runtime.h>
#include <hip/hip_bf16.h>

// ---------------------------------------------------------------------------
// Types for CDNA5 WMMA (gfx1250, wave32)
// ---------------------------------------------------------------------------
typedef __bf16 bf16;
typedef bf16  v16bf __attribute__((ext_vector_type(16)));
typedef float v8f   __attribute__((ext_vector_type(8)));

static_assert(sizeof(v16bf) == 32, "v16bf must be 32 bytes");

union FragA {
    v16bf v;
    uint4 u[2];
};

__device__ __forceinline__ v8f wmma_bf16(v16bf a, v16bf b, v8f c) {
    // D = A(16x32 bf16) x B(32x16 bf16) + C(16x16 f32)
    return __builtin_amdgcn_wmma_f32_16x16x32_bf16(
        /*neg_a=*/false, a, /*neg_b=*/false, b,
        /*c_mod=*/(short)0, c, /*reuse_a=*/false, /*reuse_b=*/false);
}

// CDNA5 async global->LDS copy (16B per lane), ASYNCcnt tracked.
__device__ __forceinline__ void async_ld_b128(unsigned lds_byte_off, const void* gaddr) {
    asm volatile("global_load_async_to_lds_b128 %0, %1, off"
                 :
                 : "v"(lds_byte_off), "v"(gaddr)
                 : "memory");
}
__device__ __forceinline__ void async_wait0() {
    asm volatile("s_wait_asynccnt 0x0" ::: "memory");
}
// Low 32 bits of a generic LDS address are the wave-relative LDS byte address.
__device__ __forceinline__ unsigned lds_off(const void* p) {
    return (unsigned)(unsigned long long)p;
}

// Problem constants
#define Bq   4
#define Sq   4096
#define Dq   1024
#define Hq   16
#define Kq   64
#define Mq   256
#define FFq  4096
#define BH   (Bq * Hq)
#define BS   (Bq * Sq)

// ---------------------------------------------------------------------------
// Elementwise conversion kernels
// ---------------------------------------------------------------------------
__global__ void cvt_bf16_k(const float* __restrict__ in, bf16* __restrict__ out, long n) {
    long i = (long)blockIdx.x * blockDim.x + threadIdx.x;
    long stride = (long)gridDim.x * blockDim.x;
    for (; i < n; i += stride) out[i] = (bf16)in[i];
}

// Interleave wq|wk|wv into one [D, 3D] bf16 matrix (each is [D, D] flat).
__global__ void cvt_qkvw_k(const float* __restrict__ wq, const float* __restrict__ wk,
                           const float* __restrict__ wv, bf16* __restrict__ out) {
    const long n = (long)Dq * 3 * Dq;
    long i = (long)blockIdx.x * blockDim.x + threadIdx.x;
    long stride = (long)gridDim.x * blockDim.x;
    for (; i < n; i += stride) {
        int r = (int)(i / (3 * Dq));
        int c = (int)(i % (3 * Dq));
        int sel = c >> 10;
        int cc  = c & 1023;
        const float* src = (sel == 0) ? wq : (sel == 1) ? wk : wv;
        out[i] = (bf16)src[(long)r * Dq + cc];
    }
}

// ---------------------------------------------------------------------------
// Generic WMMA GEMM: 128xTN tile, K step 64 (two wmma k-steps per stage),
// 8 waves (4 row x 2 col). Epilogue selected at compile time.
// A tiles staged via async global->LDS (except EPI_DIV which folds den into
// the register path); B tiles staged transposed through VGPR scatter.
// ---------------------------------------------------------------------------
constexpr int EPI_F32  = 0;  // outf = acc (+bias) (+resid)
constexpr int EPI_GELU = 1;  // outb = bf16(gelu(acc + bias))
constexpr int EPI_QKV  = 2;  // scatter scaled bf16 into q/k/v [BH,S,K]
constexpr int EPI_DIV  = 3;  // attn scatter: bf16(acc / den) with den folded in

struct EpiArgs {
    const float* bias;
    const float* resid;
    float*       outf;
    bf16*        outb;
    bf16*        outq;
    bf16*        outk;
    bf16*        outv;
    const float* zvec;
    bf16*        attn;
    float        scale;
    int          ldc;
};

template <int TN, int EPI>
__global__ __launch_bounds__(256) void gemm_k(const bf16* __restrict__ A, long strA, int lda,
                                              const bf16* __restrict__ B, long strB, int ldb,
                                              int Kdim, EpiArgs e) {
    constexpr int TM = 128, KT = 64, LD = 72;  // LD: padded LDS stride (halves)
    constexpr int FN = TN / 32;                // col fragments per wave

    __shared__ bf16  sA[TM * LD];
    __shared__ bf16  sB[TN * LD];
    __shared__ float sden[TM][2];

    const int tid  = threadIdx.x;
    const int lane = tid & 31;
    const int wv   = tid >> 5;
    const int wr   = wv & 3;   // 4 row-waves, 32 rows each
    const int wc   = wv >> 2;  // 2 col-waves, TN/2 cols each

    const long rowBlock = (long)blockIdx.y * TM;
    const int  colBlock = blockIdx.x * TN;
    const bf16* Ab = A + (long)blockIdx.z * strA;
    const bf16* Bb = B + (long)blockIdx.z * strB;
    const float* zb = (EPI == EPI_DIV) ? (e.zvec + (long)blockIdx.z * Mq) : nullptr;
    const unsigned sA_base = lds_off(&sA[0]);

    v8f acc[2][FN] = {};
    float denp = 0.f;

    for (int kt = 0; kt < Kdim; kt += KT) {
        __syncthreads();
        // ---- Stage A tile: TM x 64 halves, row-major. 4 chunks of 8/thread.
        #pragma unroll
        for (int i = 0; i < 4; ++i) {
            int c  = tid * 4 + i;
            int r  = c >> 3;            // 8 chunks per row
            int kp = (c & 7) << 3;
            const bf16* g = Ab + (rowBlock + r) * (long)lda + kt + kp;
            if constexpr (EPI == EPI_DIV) {
                uint4 d = *(const uint4*)g;
                *(uint4*)&sA[r * LD + kp] = d;
                const bf16* h = (const bf16*)&d;
                #pragma unroll
                for (int j = 0; j < 8; ++j) denp += (float)h[j] * zb[kt + kp + j];
            } else {
                async_ld_b128(sA_base + (unsigned)(r * LD + kp) * 2u, g);
            }
        }
        // ---- Stage B tile transposed: LDS[n][k] from row-major B.
        constexpr int CHB = TN / 32;  // chunks per thread (128->4, 64->2)
        #pragma unroll
        for (int i = 0; i < CHB; ++i) {
            int c  = tid * CHB + i;
            int kr = c / (TN / 8);
            int np = (c % (TN / 8)) * 8;
            uint4 d = *(const uint4*)(Bb + (long)(kt + kr) * ldb + colBlock + np);
            const bf16* h = (const bf16*)&d;
            #pragma unroll
            for (int j = 0; j < 8; ++j) sB[(np + j) * LD + kr] = h[j];
            if (i == 0 && kt + KT < Kdim)
                __builtin_prefetch((const void*)(Bb + (long)(kt + KT + kr) * ldb + colBlock + np), 0, 1);
        }
        if constexpr (EPI != EPI_DIV) async_wait0();
        __syncthreads();

        const int l15 = lane & 15, lh = lane >> 4;
        const int r0 = wr * 32, cB = wc * (TN / 2);
        #pragma unroll
        for (int kk = 0; kk < KT; kk += 32) {
            FragA a[2];
            #pragma unroll
            for (int ri = 0; ri < 2; ++ri) {
                int rr = r0 + ri * 16 + l15;
                a[ri].u[0] = *(const uint4*)&sA[rr * LD + kk + 8 * lh];
                a[ri].u[1] = *(const uint4*)&sA[rr * LD + kk + 16 + 8 * lh];
            }
            #pragma unroll
            for (int ci = 0; ci < FN; ++ci) {
                int nn = cB + ci * 16 + l15;
                FragA b;
                b.u[0] = *(const uint4*)&sB[nn * LD + kk + 8 * lh];
                b.u[1] = *(const uint4*)&sB[nn * LD + kk + 16 + 8 * lh];
                #pragma unroll
                for (int ri = 0; ri < 2; ++ri)
                    acc[ri][ci] = wmma_bf16(a[ri].v, b.v, acc[ri][ci]);
            }
        }
    }

    if constexpr (EPI == EPI_DIV) {
        sden[tid >> 1][tid & 1] = denp;  // thread t covered local row t/2
        __syncthreads();
    }

    const int l15 = lane & 15, lh = lane >> 4;
    const int r0 = wr * 32, cB = wc * (TN / 2);
    #pragma unroll
    for (int ri = 0; ri < 2; ++ri) {
        #pragma unroll
        for (int ci = 0; ci < FN; ++ci) {
            const int n = colBlock + cB + ci * 16 + l15;
            #pragma unroll
            for (int j = 0; j < 8; ++j) {
                const int  lr = r0 + ri * 16 + j + 8 * lh;
                const long m  = rowBlock + lr;
                float v = acc[ri][ci][j];
                if constexpr (EPI == EPI_F32) {
                    if (e.bias)  v += e.bias[n];
                    if (e.resid) v += e.resid[m * e.ldc + n];
                    e.outf[m * e.ldc + n] = v;
                } else if constexpr (EPI == EPI_GELU) {
                    v += e.bias[n];
                    float g = 0.5f * v * (1.f + erff(v * 0.70710678118f));
                    e.outb[m * e.ldc + n] = (bf16)g;
                } else if constexpr (EPI == EPI_QKV) {
                    int sel = n >> 10;       // 0:q 1:k 2:v
                    int c1  = n & 1023;
                    int h   = c1 >> 6, k = c1 & 63;
                    int b   = (int)(m >> 12);
                    int s   = (int)(m & 4095);
                    long idx = (((long)(b * Hq + h)) * Sq + s) * Kq + k;
                    float val = (sel == 2) ? v : v * e.scale;
                    bf16* dst = (sel == 0) ? e.outq : (sel == 1) ? e.outk : e.outv;
                    dst[idx] = (bf16)val;
                } else {  // EPI_DIV
                    float dden = sden[lr][0] + sden[lr][1];
                    float val  = v / dden;
                    int bh = blockIdx.z;
                    int b = bh >> 4, h = bh & 15;
                    long idx = ((long)b * Sq + m) * Dq + h * Kq + n;
                    e.attn[idx] = (bf16)val;
                }
            }
        }
    }
}

// ---------------------------------------------------------------------------
// FAVOR+ feature kernel: per (bh, 32-row block), proj = u @ omega^T over all
// M=256 columns in one K=64 pass; phi = exp(proj - 0.5|u|^2 - rowmax)/16 + 1e-6.
// sq folded into A staging; omega staged via async-to-LDS; the f32 reduction
// tile aliases the B tile through a union.
// ---------------------------------------------------------------------------
union PhiShared {
    bf16  sB[256 * 72];
    float ct[32 * 258];
};

__global__ __launch_bounds__(256) void phi_kernel(const bf16* __restrict__ U,   // [BH,S,64]
                                                  const bf16* __restrict__ om,  // [256,64]
                                                  bf16* __restrict__ phi) {     // [BH,S,256]
    constexpr int LD = 72;
    __shared__ bf16      sA[32 * LD];
    __shared__ PhiShared shb;
    __shared__ float ssq[32][8];
    __shared__ float smax[32][8];
    __shared__ float sstab[32];
    __shared__ float ssqf[32];

    const int tid = threadIdx.x, lane = tid & 31, wv = tid >> 5;
    const int bh = blockIdx.y;
    const int rowBlock = blockIdx.x * 32;
    const bf16* Ab = U + (long)bh * Sq * Kq + (long)rowBlock * Kq;
    const unsigned sB_base = lds_off(&shb.sB[0]);

    // A: 32 x 64 halves, 1 chunk of 8 per thread; fold 0.5*|u|^2.
    {
        int r = tid >> 3, kp = (tid & 7) << 3;
        uint4 d = *(const uint4*)(Ab + (long)r * Kq + kp);
        *(uint4*)&sA[r * LD + kp] = d;
        const bf16* h = (const bf16*)&d;
        float s = 0.f;
        #pragma unroll
        for (int j = 0; j < 8; ++j) { float f = (float)h[j]; s += f * f; }
        ssq[r][tid & 7] = 0.5f * s;
    }
    // B: omega rows are already B^T rows -> contiguous async copy, 8 chunks.
    #pragma unroll
    for (int i = 0; i < 8; ++i) {
        int c = tid * 8 + i;
        int n = c >> 3, kp = (c & 7) << 3;
        async_ld_b128(sB_base + (unsigned)(n * LD + kp) * 2u, om + (long)n * Kq + kp);
    }
    async_wait0();
    __syncthreads();

    v8f acc[2][2] = {};
    const int l15 = lane & 15, lh = lane >> 4;
    const int cB = wv * 32;
    #pragma unroll
    for (int kk = 0; kk < 64; kk += 32) {
        FragA a[2];
        #pragma unroll
        for (int ri = 0; ri < 2; ++ri) {
            int rr = ri * 16 + l15;
            a[ri].u[0] = *(const uint4*)&sA[rr * LD + kk + 8 * lh];
            a[ri].u[1] = *(const uint4*)&sA[rr * LD + kk + 16 + 8 * lh];
        }
        #pragma unroll
        for (int ci = 0; ci < 2; ++ci) {
            int nn = cB + ci * 16 + l15;
            FragA b;
            b.u[0] = *(const uint4*)&shb.sB[nn * LD + kk + 8 * lh];
            b.u[1] = *(const uint4*)&shb.sB[nn * LD + kk + 16 + 8 * lh];
            #pragma unroll
            for (int ri = 0; ri < 2; ++ri)
                acc[ri][ci] = wmma_bf16(a[ri].v, b.v, acc[ri][ci]);
        }
    }
    __syncthreads();  // all waves done reading sB before ct overwrites it
    #pragma unroll
    for (int ri = 0; ri < 2; ++ri)
        #pragma unroll
        for (int ci = 0; ci < 2; ++ci) {
            int n = cB + ci * 16 + l15;
            #pragma unroll
            for (int j = 0; j < 8; ++j) {
                int r = ri * 16 + j + 8 * lh;
                shb.ct[r * 258 + n] = acc[ri][ci][j];
            }
        }
    __syncthreads();
    {   // per-row max over 256 cols: 8 segments of 32
        int r = tid >> 3, seg = (tid & 7) * 32;
        float m = -3.0e38f;
        for (int j = 0; j < 32; ++j) m = fmaxf(m, shb.ct[r * 258 + seg + j]);
        smax[r][tid & 7] = m;
    }
    __syncthreads();
    if (tid < 32) {
        float m = smax[tid][0];
        float q = ssq[tid][0];
        #pragma unroll
        for (int j = 1; j < 8; ++j) { m = fmaxf(m, smax[tid][j]); q += ssq[tid][j]; }
        sstab[tid] = m;
        ssqf[tid]  = q;
    }
    __syncthreads();
    bf16* out = phi + (long)bh * Sq * Mq + (long)rowBlock * Mq;
    for (int idx = tid; idx < 32 * 256; idx += 256) {
        int r = idx >> 8, n = idx & 255;
        float v = expf(shb.ct[r * 258 + n] - ssqf[r] - sstab[r]) * 0.0625f + 1e-6f;
        out[(long)r * Mq + n] = (bf16)v;
    }
}

// ---------------------------------------------------------------------------
// kv kernel: per bh, kv[256,64] = phik^T[256,S] @ v[S,64]; transposed staging
// ---------------------------------------------------------------------------
__global__ __launch_bounds__(256) void kv_kernel(const bf16* __restrict__ phik,  // [BH,S,256]
                                                 const bf16* __restrict__ vmat,  // [BH,S,64]
                                                 bf16* __restrict__ kv) {        // [BH,256,64]
    constexpr int LD = 72, KT = 64;
    __shared__ bf16 sA[256 * LD];
    __shared__ bf16 sB[64 * LD];

    const int tid = threadIdx.x, lane = tid & 31, wv = tid >> 5;
    const int bh = blockIdx.x;
    const bf16* Pb = phik + (long)bh * Sq * Mq;
    const bf16* Vb = vmat + (long)bh * Sq * Kq;

    v8f acc[2][4] = {};

    for (int kt = 0; kt < Sq; kt += KT) {
        __syncthreads();
        // A: 256 m x 64 s, transposed scatter; 8 chunks of 8 per thread
        #pragma unroll
        for (int i = 0; i < 8; ++i) {
            int c = tid * 8 + i;
            int sl = c >> 5, mp = (c & 31) << 3;
            uint4 d = *(const uint4*)(Pb + (long)(kt + sl) * Mq + mp);
            const bf16* h = (const bf16*)&d;
            #pragma unroll
            for (int j = 0; j < 8; ++j) sA[(mp + j) * LD + sl] = h[j];
            if (i == 0 && kt + KT < Sq)
                __builtin_prefetch((const void*)(Pb + (long)(kt + KT + sl) * Mq + mp), 0, 1);
        }
        // B: 64 s x 64 k, transposed scatter; 2 chunks per thread
        #pragma unroll
        for (int i = 0; i < 2; ++i) {
            int c = tid * 2 + i;
            int sl = c >> 3, kp = (c & 7) << 3;
            uint4 d = *(const uint4*)(Vb + (long)(kt + sl) * Kq + kp);
            const bf16* h = (const bf16*)&d;
            #pragma unroll
            for (int j = 0; j < 8; ++j) sB[(kp + j) * LD + sl] = h[j];
        }
        __syncthreads();

        const int l15 = lane & 15, lh = lane >> 4;
        const int r0 = wv * 32;
        #pragma unroll
        for (int kk = 0; kk < KT; kk += 32) {
            FragA a[2];
            #pragma unroll
            for (int ri = 0; ri < 2; ++ri) {
                int rr = r0 + ri * 16 + l15;
                a[ri].u[0] = *(const uint4*)&sA[rr * LD + kk + 8 * lh];
                a[ri].u[1] = *(const uint4*)&sA[rr * LD + kk + 16 + 8 * lh];
            }
            #pragma unroll
            for (int ci = 0; ci < 4; ++ci) {
                int nn = ci * 16 + l15;
                FragA b;
                b.u[0] = *(const uint4*)&sB[nn * LD + kk + 8 * lh];
                b.u[1] = *(const uint4*)&sB[nn * LD + kk + 16 + 8 * lh];
                #pragma unroll
                for (int ri = 0; ri < 2; ++ri)
                    acc[ri][ci] = wmma_bf16(a[ri].v, b.v, acc[ri][ci]);
            }
        }
    }

    const int l15 = lane & 15, lh = lane >> 4, r0 = wv * 32;
    bf16* outp = kv + (long)bh * Mq * Kq;
    #pragma unroll
    for (int ri = 0; ri < 2; ++ri)
        #pragma unroll
        for (int ci = 0; ci < 4; ++ci) {
            int n = ci * 16 + l15;
            #pragma unroll
            for (int j = 0; j < 8; ++j) {
                int m = r0 + ri * 16 + j + 8 * lh;
                outp[(long)m * Kq + n] = (bf16)acc[ri][ci][j];
            }
        }
}

// z[bh,m] = sum_s phik[bh,s,m]
__global__ __launch_bounds__(256) void zsum_k(const bf16* __restrict__ phik, float* __restrict__ z) {
    int bh = blockIdx.x, m = threadIdx.x;
    const bf16* p = phik + (long)bh * Sq * Mq + m;
    float a = 0.f;
    for (int s = 0; s < Sq; ++s) a += (float)p[(long)s * Mq];
    z[(long)bh * Mq + m] = a;
}

// ---------------------------------------------------------------------------
// LayerNorm over D=1024; optional bf16 secondary output
// ---------------------------------------------------------------------------
__global__ __launch_bounds__(256) void layernorm_k(const float* __restrict__ in,
                                                   const float* __restrict__ g,
                                                   const float* __restrict__ bb,
                                                   float* __restrict__ outf,
                                                   bf16* outb, int D) {
    const long row = blockIdx.x;
    const float* xr = in + row * D;
    __shared__ float red[256];
    const int tid = threadIdx.x;

    float s = 0.f;
    for (int i = tid; i < D; i += 256) s += xr[i];
    red[tid] = s; __syncthreads();
    for (int o = 128; o; o >>= 1) { if (tid < o) red[tid] += red[tid + o]; __syncthreads(); }
    const float mu = red[0] * (1.0f / D);
    __syncthreads();

    float vs = 0.f;
    for (int i = tid; i < D; i += 256) { float d = xr[i] - mu; vs += d * d; }
    red[tid] = vs; __syncthreads();
    for (int o = 128; o; o >>= 1) { if (tid < o) red[tid] += red[tid + o]; __syncthreads(); }
    const float rstd = rsqrtf(red[0] * (1.0f / D) + 1e-6f);

    for (int i = tid; i < D; i += 256) {
        float y = (xr[i] - mu) * rstd * g[i] + bb[i];
        outf[row * D + i] = y;
        if (outb) outb[row * D + i] = (bf16)y;
    }
}

// ---------------------------------------------------------------------------
// Host-side orchestration
// ---------------------------------------------------------------------------
extern "C" void kernel_launch(void* const* d_in, const int* in_sizes, int n_in,
                              void* d_out, int out_size, void* d_ws, size_t ws_size,
                              hipStream_t stream) {
    (void)in_sizes; (void)n_in; (void)out_size; (void)ws_size;
    const float* x     = (const float*)d_in[0];
    const float* wq    = (const float*)d_in[1];
    const float* wk    = (const float*)d_in[2];
    const float* wv    = (const float*)d_in[3];
    const float* wo    = (const float*)d_in[4];
    const float* omega = (const float*)d_in[5];
    const float* ln1g  = (const float*)d_in[6];
    const float* ln1b_ = (const float*)d_in[7];
    const float* w1    = (const float*)d_in[8];
    const float* b1    = (const float*)d_in[9];
    const float* w2    = (const float*)d_in[10];
    const float* b2    = (const float*)d_in[11];
    const float* ln2g  = (const float*)d_in[12];
    const float* ln2b_ = (const float*)d_in[13];

    char* ws = (char*)d_ws;
    size_t cur = 0;
    auto take = [&](size_t bytes) -> char* {
        char* p = ws + cur;
        cur += (bytes + 255) & ~(size_t)255;
        return p;
    };

    // Persistent buffers
    bf16*  xb    = (bf16*)take((size_t)BS * Dq * 2);
    bf16*  wqkvb = (bf16*)take((size_t)Dq * 3 * Dq * 2);
    bf16*  omb   = (bf16*)take((size_t)Mq * Kq * 2);
    bf16*  wob   = (bf16*)take((size_t)Dq * Dq * 2);
    bf16*  w1b   = (bf16*)take((size_t)Dq * FFq * 2);
    bf16*  w2b   = (bf16*)take((size_t)FFq * Dq * 2);
    bf16*  kvb   = (bf16*)take((size_t)BH * Mq * Kq * 2);
    float* zb    = (float*)take((size_t)BH * Mq * 4);
    bf16*  attnb = (bf16*)take((size_t)BS * Dq * 2);
    // Reusable regions (lifetime-based aliasing)
    char* R5 = take((size_t)BH * Sq * Mq * 2);  // phiq -> h1b
    char* R6 = take((size_t)BH * Sq * Mq * 2);  // phik -> (ln1f | ln1b)
    char* R7 = take((size_t)3 * BS * Dq * 2);   // q|k|v -> y1/y2
    bf16*  phiq  = (bf16*)R5;
    bf16*  h1b   = (bf16*)R5;
    bf16*  phik  = (bf16*)R6;
    float* ln1f  = (float*)R6;
    bf16*  ln1bq = (bf16*)(R6 + (size_t)BS * Dq * 4);
    bf16*  qb = (bf16*)R7;
    bf16*  kb = (bf16*)(R7 + (size_t)BS * Dq * 2);
    bf16*  vb = (bf16*)(R7 + (size_t)2 * BS * Dq * 2);
    float* y1 = (float*)R7;
    float* y2 = y1;

    // ---- 0) fp32 -> bf16 conversion ----
    cvt_bf16_k<<<2048, 256, 0, stream>>>(x, xb, (long)BS * Dq);
    cvt_qkvw_k<<<2048, 256, 0, stream>>>(wq, wk, wv, wqkvb);
    cvt_bf16_k<<<64, 256, 0, stream>>>(omega, omb, (long)Mq * Kq);
    cvt_bf16_k<<<1024, 256, 0, stream>>>(wo, wob, (long)Dq * Dq);
    cvt_bf16_k<<<2048, 256, 0, stream>>>(w1, w1b, (long)Dq * FFq);
    cvt_bf16_k<<<2048, 256, 0, stream>>>(w2, w2b, (long)FFq * Dq);

    // ---- 1) QKV projection ----
    {
        EpiArgs e{};
        e.outq = qb; e.outk = kb; e.outv = vb;
        e.scale = 0.35355339059f;  // K^-0.25
        gemm_k<128, EPI_QKV><<<dim3(3 * Dq / 128, BS / 128, 1), 256, 0, stream>>>(
            xb, 0, Dq, wqkvb, 0, 3 * Dq, Dq, e);
    }

    // ---- 2) FAVOR+ features ----
    phi_kernel<<<dim3(Sq / 32, BH), 256, 0, stream>>>(qb, omb, phiq);
    phi_kernel<<<dim3(Sq / 32, BH), 256, 0, stream>>>(kb, omb, phik);

    // ---- 3) z = sum_s phik ----
    zsum_k<<<BH, 256, 0, stream>>>(phik, zb);

    // ---- 4) kv = phik^T @ v ----
    kv_kernel<<<BH, 256, 0, stream>>>(phik, vb, kvb);

    // ---- 5) attn = (phiq @ kv) / (phiq . z) ----
    {
        EpiArgs e{};
        e.zvec = zb; e.attn = attnb;
        gemm_k<64, EPI_DIV><<<dim3(1, Sq / 128, BH), 256, 0, stream>>>(
            phiq, (long)Sq * Mq, Mq, kvb, (long)Mq * Kq, Kq, Mq, e);
    }

    // ---- 6) y1 = x + attn @ wo ----
    {
        EpiArgs e{};
        e.outf = y1; e.resid = x; e.ldc = Dq;
        gemm_k<128, EPI_F32><<<dim3(Dq / 128, BS / 128, 1), 256, 0, stream>>>(
            attnb, 0, Dq, wob, 0, Dq, Dq, e);
    }

    // ---- 7) LN1 ----
    layernorm_k<<<BS, 256, 0, stream>>>(y1, ln1g, ln1b_, ln1f, ln1bq, Dq);

    // ---- 8) FFN1: gelu ----
    {
        EpiArgs e{};
        e.outb = h1b; e.bias = b1; e.ldc = FFq;
        gemm_k<128, EPI_GELU><<<dim3(FFq / 128, BS / 128, 1), 256, 0, stream>>>(
            ln1bq, 0, Dq, w1b, 0, FFq, Dq, e);
    }

    // ---- 9) FFN2 + residual ----
    {
        EpiArgs e{};
        e.outf = y2; e.bias = b2; e.resid = ln1f; e.ldc = Dq;
        gemm_k<128, EPI_F32><<<dim3(Dq / 128, BS / 128, 1), 256, 0, stream>>>(
            h1b, 0, FFq, w2b, 0, Dq, FFq, e);
    }

    // ---- 10) LN2 -> d_out ----
    layernorm_k<<<BS, 256, 0, stream>>>(y2, ln2g, ln2b_, (float*)d_out, nullptr, Dq);
}
